// SelfAttention_14285061227125
// MI455X (gfx1250) — matrix operand order repaired and checked by reading, not compiled, not run on hardware
//
#include <hip/hip_runtime.h>
#include <hip/hip_bf16.h>

// ---------------------------------------------------------------------------
// Fused self-attention for B=8, C=Co=256, N=H*W=4096 on gfx1250 (MI455X).
// Pipeline: [weights->f16] -> [QKV proj f16-WMMA] ->
//           [flash attention f16-WMMA + TDM K-staging, online softmax] ->
//           [out proj f16-WMMA].
// Workspace: Q|K|V|AO f16 (64 MB) + f16 weights (512 KB).
// ---------------------------------------------------------------------------

typedef __attribute__((ext_vector_type(16))) _Float16     v16h;
typedef __attribute__((ext_vector_type(8)))  float        v8f;
typedef __attribute__((ext_vector_type(4)))  unsigned int u32x4;
typedef __attribute__((ext_vector_type(4)))  float        f32x4;
typedef __attribute__((ext_vector_type(8)))  int          i32x8;
typedef __attribute__((ext_vector_type(4)))  int          i32x4;

union V16H { v16h v; u32x4 u[2]; };
union H8   { u32x4 u; _Float16 h[8]; };

#define BATCH 8
#define NTOK  4096
#define CH    256

// LDS row pads (multiples of 8 f16 keep 16B fragment loads aligned)
#define XS_LD 264
#define KV_LD 264
#define VT_LD 64     // Vt rows: swizzled chunks, no pad needed
#define P_LD  72

// TDM availability: ROCm 7.2 (clang-22) 5-arg builtin form (probe-verified).
#if defined(__has_builtin)
#if __has_builtin(__builtin_amdgcn_tensor_load_to_lds) && (__clang_major__ <= 22)
#define USE_TDM 1
#endif
#endif
#ifndef USE_TDM
#define USE_TDM 0
#endif

// D = A(16x32 f16) * B(32x16 f16) + C(16x16 f32)
__device__ inline v8f wmma_f16(v16h a, v16h b, v8f c) {
  return __builtin_amdgcn_wmma_f32_16x16x32_f16(
      false, a, false, b, (short)0, c, false, false);
}

// A-fragment (16x32 f16): lane = row (lane&15); elems i -> K = 2*kb8+i,
// kb8 = (lane>>4)*8.
__device__ inline v16h load_afrag(const _Float16* row, int kb8) {
  V16H r;
  r.u[0] = *(const u32x4*)(row + 2 * kb8);
  r.u[1] = *(const u32x4*)(row + 2 * kb8 + 8);
  return r.v;
}

// B-fragment (32x16 f16): lane = col (lane&15); elems i -> K = kb16+i.
__device__ inline v16h load_bfrag16(const _Float16* p) {
  V16H r;
  r.u[0] = *(const u32x4*)(p);
  r.u[1] = *(const u32x4*)(p + 8);
  return r.v;
}

// Swizzled 8-f16 chunk position for transposed-V LDS tile (bank decorrelation).
__device__ inline int vchunk(int o, int cj) { return (cj + o + (o >> 3)) & 7; }

// ---------------------------------------------------------------------------
// Kernel 0: convert the four 256x256 f32 weight matrices to f16 (one shot).
// ---------------------------------------------------------------------------
__global__ __launch_bounds__(256) void cvt_weights_kernel(
    const float* __restrict__ wq, const float* __restrict__ wk,
    const float* __restrict__ wv, const float* __restrict__ wo,
    _Float16* __restrict__ out)   // wqh | wkh | wvh | woh, each 65536 elems
{
  const int gid = blockIdx.x * 256 + threadIdx.x;   // 0..16383
  const float* src[4] = {wq, wk, wv, wo};
#pragma unroll
  for (int m = 0; m < 4; ++m) {
    f32x4 v = *(const f32x4*)(src[m] + (size_t)gid * 4);
    _Float16* d = out + (size_t)m * 65536 + (size_t)gid * 4;
#pragma unroll
    for (int e = 0; e < 4; ++e) d[e] = (_Float16)v[e];
  }
}

// ---------------------------------------------------------------------------
// Kernel 1: Q/K/V = xf * W^T + b, output f16 [b][n][o].
// Outputs staged in LDS, then copied out with coalesced b128 stores.
// Grid: BATCH * (NTOK/32) = 1024 blocks of 256 threads (8 waves).
// ---------------------------------------------------------------------------
__global__ __launch_bounds__(256) void qkv_proj_kernel(
    const float* __restrict__ x,
    const _Float16* __restrict__ wqh, const _Float16* __restrict__ wkh,
    const _Float16* __restrict__ wvh,
    const float* __restrict__ bq, const float* __restrict__ bk,
    const float* __restrict__ bv,
    _Float16* __restrict__ Q, _Float16* __restrict__ K, _Float16* __restrict__ V)
{
  __shared__ _Float16 xs[32 * XS_LD];        // xs[r][c] = x[b][c][n0+r]
  __shared__ _Float16 outs[3 * 32 * XS_LD];  // q|k|v staged rows

  const int bid = blockIdx.x;
  const int b   = bid >> 7;
  const int n0  = (bid & 127) << 5;
  const int t   = threadIdx.x;

  // Transpose-load x tile: consecutive threads -> consecutive n (coalesced).
  {
    const int r  = t & 31;
    const int c8 = t >> 5;
    const float* xb = x + (size_t)b * CH * NTOK;
    for (int p = 0; p < 32; ++p) {
      const int c = p * 8 + c8;
      xs[r * XS_LD + c] = (_Float16)xb[(size_t)c * NTOK + n0 + r];
    }
  }
  __syncthreads();

  const int lane = t & 31, wave = t >> 5;
  const int cl   = lane & 15;
  const int half = lane >> 4;
  const int kb8 = half * 8, kb16 = half * 16, rbase = half * 8;

  // 96 tiles: 2 row-blocks x {q,k,v} x 16 col-tiles; 12 contiguous per wave.
  for (int i = 0; i < 12; ++i) {
    const int tile = wave * 12 + i;
    const int rb  = tile / 48;
    const int rem = tile % 48;
    const int m   = rem >> 4;
    const int ot  = rem & 15;
    const _Float16* wsel = (m == 0) ? wqh : (m == 1) ? wkh : wvh;
    const float*    bsel = (m == 0) ? bq  : (m == 1) ? bk  : bv;
    const int o0 = ot * 16;

    v8f acc = {};
    const _Float16* arow = &xs[(rb * 16 + cl) * XS_LD];
    const _Float16* wrow = wsel + (size_t)(o0 + cl) * CH;
#pragma unroll
    for (int kt = 0; kt < 8; ++kt) {
      v16h a  = load_afrag(arow + kt * 32, kb8);
      v16h bf = load_bfrag16(wrow + kt * 32 + kb16);
      acc = wmma_f16(a, bf, acc);
    }
    const float bias = bsel[o0 + cl];
    _Float16* drow = &outs[(m * 32 + rb * 16 + rbase) * XS_LD + o0 + cl];
#pragma unroll
    for (int g = 0; g < 8; ++g)
      drow[g * XS_LD] = (_Float16)(acc[g] + bias);
  }
  __syncthreads();

  // Coalesced copy-out: 3 matrices x 32 rows x 32 chunks of 8 f16.
  for (int i = t; i < 3 * 32 * 32; i += 256) {
    const int m   = i >> 10;
    const int rem = i & 1023;
    const int r   = rem >> 5, cc = rem & 31;
    u32x4 d = *(const u32x4*)&outs[(m * 32 + r) * XS_LD + cc * 8];
    _Float16* dsel = (m == 0) ? Q : (m == 1) ? K : V;
    *(u32x4*)(dsel + ((size_t)b * NTOK + n0 + r) * CH + cc * 8) = d;
  }
}

// ---------------------------------------------------------------------------
// Kernel 2: flash attention. 128 q-rows per block (8 waves x 16-row tile),
// KV blocks of 64. K staged via TDM (tensor_load_to_lds, HW row-padding to the
// 264-f16 pitch) or manual fallback; V staged TRANSPOSED (Vt[o][j], swizzled
// chunks) via in-register 8x8 f16 transpose. AO written via LDS, coalesced.
// Grid: BATCH * (NTOK/128) = 256 blocks. Dynamic LDS ~ 83 KB.
// ---------------------------------------------------------------------------
__global__ __launch_bounds__(256) void flash_attn_kernel(
    const _Float16* __restrict__ Q, const _Float16* __restrict__ K,
    const _Float16* __restrict__ V, _Float16* __restrict__ AO)
{
  extern __shared__ _Float16 smem[];
  _Float16* Ks = smem;                  // 64 x KV_LD   (natural [j][o])
  _Float16* Vt = Ks + 64 * KV_LD;       // 256 x VT_LD  (transposed [o][j])
  _Float16* Ps = Vt + 256 * VT_LD;      // 8 waves x 16 x P_LD

  const int bid = blockIdx.x;
  const int b   = bid >> 5;
  const int q0  = (bid & 31) << 7;
  const int t = threadIdx.x, lane = t & 31, wave = t >> 5;
  const int cl = lane & 15, half = lane >> 4;
  const int kb8 = half * 8, kb16 = half * 16, rbase = half * 8;

  const _Float16* Qb = Q + (size_t)b * NTOK * CH;
  const _Float16* Kb = K + (size_t)b * NTOK * CH;
  const _Float16* Vb = V + (size_t)b * NTOK * CH;

  float m_r[8], l_r[8];
  v8f oacc[16] = {};
#pragma unroll
  for (int g = 0; g < 8; ++g) { m_r[g] = -1e30f; l_r[g] = 0.f; }

  const _Float16* qrow = Qb + (size_t)(q0 + wave * 16 + cl) * CH;
  _Float16* pw = Ps + wave * 16 * P_LD;

  const int jb = t >> 5;   // V-transpose staging: j-block 0..7
  const int oc = t & 31;   // o-chunk 0..31

#if USE_TDM
  // LDS byte offset of Ks (flat LDS pointers carry the offset in low 32 bits)
  const unsigned lds_ks = (unsigned)(size_t)Ks;
#endif

  for (int j0 = 0; j0 < NTOK; j0 += 64) {
    // ---- stage K ----
#if USE_TDM
    if (wave == 0) {
      const unsigned long long ga =
          (unsigned long long)(size_t)(Kb + (size_t)j0 * CH);
      u32x4 g0;
      g0[0] = 1u;                                     // count=1, user mode
      g0[1] = lds_ks;                                 // lds_addr
      g0[2] = (unsigned)ga;                           // global_addr[31:0]
      g0[3] = (unsigned)((ga >> 32) & 0x1FFFFFFu)     // global_addr[56:32]
              | (2u << 30);                           // type=2 (image)
      i32x8 g1;
      g1[0] = (int)((1u << 16)      // data_size = 2 bytes
                  | (1u << 20)      // pad_enable
                  | (6u << 22)      // pad_interval: 128 DWORDs (one 512B row)
                  | (3u << 25));    // pad_amount: 4 DWORDs (16B) -> 264 pitch
      g1[1] = (int)(256u << 16);    // tensor_dim0 = 256
      g1[2] = (int)(64u << 16);     // tensor_dim1 = 64
      g1[3] = (int)(256u << 16);    // tile_dim0 = 256
      g1[4] = 64;                   // tile_dim1 = 64 (tile_dim2 = 0)
      g1[5] = 256;                  // tensor_dim0_stride = 256
      g1[6] = 0; g1[7] = 0;
      const i32x4 z4 = {0, 0, 0, 0};
      __builtin_amdgcn_tensor_load_to_lds(g0, g1, z4, z4, 0);
      __builtin_amdgcn_s_wait_tensorcnt(0);
    }
#else
    {
      const u32x4* kg = (const u32x4*)(Kb + (size_t)j0 * CH);
      for (int i = t; i < 64 * 32; i += 256) {
        const int r = i >> 5, cc = i & 31;
        *(u32x4*)&Ks[r * KV_LD + cc * 8] = kg[i];
      }
    }
#endif
    // ---- stage V transposed (in-register 8x8 f16 transpose) ----
    {
      const u32x4* vg = (const u32x4*)(Vb + (size_t)j0 * CH);
      H8 in[8], o8;
#pragma unroll
      for (int r = 0; r < 8; ++r) in[r].u = vg[(jb * 8 + r) * 32 + oc];
#pragma unroll
      for (int e = 0; e < 8; ++e) {
#pragma unroll
        for (int r = 0; r < 8; ++r) o8.h[r] = in[r].h[e];
        const int o = oc * 8 + e;
        *(u32x4*)&Vt[o * VT_LD + vchunk(o, jb) * 8] = o8.u;
      }
    }
    // prefetch next KV block into GL2 while this one is consumed
    if (j0 + 64 < NTOK) {
      __builtin_prefetch(Kb + (size_t)(j0 + 64) * CH + t * 64, 0, 1);
      __builtin_prefetch(Vb + (size_t)(j0 + 64) * CH + t * 64, 0, 1);
    }
    __syncthreads();

    // ---- S(16x64) = Q(16x256) x K^T(256x64) ----
    v8f s[4] = {};
#pragma unroll
    for (int kt = 0; kt < 8; ++kt) {
      v16h a = load_afrag(qrow + kt * 32, kb8);
#pragma unroll
      for (int jt = 0; jt < 4; ++jt) {
        v16h bf = load_bfrag16(&Ks[(jt * 16 + cl) * KV_LD + kt * 32 + kb16]);
        s[jt] = wmma_f16(a, bf, s[jt]);
      }
    }

    // ---- online softmax (row's 16 cols live in one 16-lane group) ----
    float mnew[8], scale[8], rsum[8];
#pragma unroll
    for (int g = 0; g < 8; ++g) {
      float v = fmaxf(fmaxf(s[0][g], s[1][g]), fmaxf(s[2][g], s[3][g]));
      v = fmaxf(v, __shfl_xor(v, 1));
      v = fmaxf(v, __shfl_xor(v, 2));
      v = fmaxf(v, __shfl_xor(v, 4));
      v = fmaxf(v, __shfl_xor(v, 8));
      mnew[g]  = fmaxf(m_r[g], v);
      scale[g] = __expf(m_r[g] - mnew[g]);
      rsum[g]  = 0.f;
    }
#pragma unroll
    for (int jt = 0; jt < 4; ++jt)
#pragma unroll
      for (int g = 0; g < 8; ++g) {
        const float p = __expf(s[jt][g] - mnew[g]);
        rsum[g] += p;
        pw[(rbase + g) * P_LD + jt * 16 + cl] = (_Float16)p;
      }
#pragma unroll
    for (int g = 0; g < 8; ++g) {
      float v = rsum[g];
      v += __shfl_xor(v, 1); v += __shfl_xor(v, 2);
      v += __shfl_xor(v, 4); v += __shfl_xor(v, 8);
      l_r[g] = l_r[g] * scale[g] + v;
      m_r[g] = mnew[g];
    }
#pragma unroll
    for (int ot = 0; ot < 16; ++ot)
#pragma unroll
      for (int g = 0; g < 8; ++g) oacc[ot][g] *= scale[g];

    // same-wave LDS RAW on Ps
    asm volatile("s_wait_dscnt 0" ::: "memory");

    // ---- O(16x256) += P(16x64) x V(64x256) ----
    v16h pa[2];
#pragma unroll
    for (int k2 = 0; k2 < 2; ++k2)
      pa[k2] = load_afrag(pw + cl * P_LD + k2 * 32, kb8);
#pragma unroll
    for (int ot = 0; ot < 16; ++ot) {
      const int o = ot * 16 + cl;
      const _Float16* vrow = &Vt[o * VT_LD];
#pragma unroll
      for (int k2 = 0; k2 < 2; ++k2) {
        const int jc = (k2 * 32 + kb16) >> 3;
        V16H bf;
        bf.u[0] = *(const u32x4*)(vrow + vchunk(o, jc) * 8);
        bf.u[1] = *(const u32x4*)(vrow + vchunk(o, jc + 1) * 8);
        oacc[ot] = wmma_f16(pa[k2], bf.v, oacc[ot]);
      }
    }
    __syncthreads();
  }

  // ---- normalize, stage in LDS, coalesced copy-out of AO f16 [b][n][o] ----
  float inv[8];
#pragma unroll
  for (int g = 0; g < 8; ++g) inv[g] = 1.f / l_r[g];
  _Float16* ob = smem;   // reuse: 128 rows x KV_LD pitch (fits in 83 KB)
#pragma unroll
  for (int ot = 0; ot < 16; ++ot)
#pragma unroll
    for (int g = 0; g < 8; ++g)
      ob[(wave * 16 + rbase + g) * KV_LD + ot * 16 + cl] =
          (_Float16)(oacc[ot][g] * inv[g]);
  __syncthreads();
  for (int i = t; i < 128 * 32; i += 256) {
    const int r = i >> 5, cc = i & 31;
    u32x4 d = *(const u32x4*)&ob[r * KV_LD + cc * 8];
    *(u32x4*)(AO + ((size_t)b * NTOK + q0 + r) * CH + cc * 8) = d;
  }
}

// ---------------------------------------------------------------------------
// Kernel 3: out[b][c][n] = AO[b][n][:] . wo[c][:] + bo[c], f32 output.
// Grid: BATCH * (NTOK/64) = 512 blocks of 256 threads.
// ---------------------------------------------------------------------------
__global__ __launch_bounds__(256) void out_proj_kernel(
    const _Float16* __restrict__ AO, const _Float16* __restrict__ woh,
    const float* __restrict__ bo, float* __restrict__ out)
{
  const int bid = blockIdx.x;
  const int b   = bid >> 6;
  const int n0  = (bid & 63) << 6;
  const int t = threadIdx.x, lane = t & 31, wave = t >> 5;
  const int cl = lane & 15, half = lane >> 4;
  const int kb8 = half * 8, kb16 = half * 16, rbase = half * 8;

  for (int tile = wave; tile < 64; tile += 8) {
    const int rb = tile >> 4;
    const int ct = tile & 15;
    v8f acc = {};
    const _Float16* arow = AO + ((size_t)b * NTOK + n0 + rb * 16 + cl) * CH;
    const _Float16* wrow = woh + (size_t)(ct * 16 + cl) * CH;
#pragma unroll
    for (int kt = 0; kt < 8; ++kt) {
      v16h a  = load_afrag(arow + kt * 32, kb8);
      v16h bf = load_bfrag16(wrow + kt * 32 + kb16);
      acc = wmma_f16(a, bf, acc);
    }
    const float bias = bo[ct * 16 + cl];
#pragma unroll
    for (int g = 0; g < 8; ++g) acc[g] += bias;
    float* op = out + ((size_t)b * CH + ct * 16 + cl) * NTOK + n0 + rb * 16 + rbase;
    f32x4 lo = { acc[0], acc[1], acc[2], acc[3] };
    f32x4 hi = { acc[4], acc[5], acc[6], acc[7] };
    *(f32x4*)(op)     = lo;
    *(f32x4*)(op + 4) = hi;
  }
}

// ---------------------------------------------------------------------------
extern "C" void kernel_launch(void* const* d_in, const int* in_sizes, int n_in,
                              void* d_out, int out_size, void* d_ws, size_t ws_size,
                              hipStream_t stream) {
  const float* x  = (const float*)d_in[0];
  const float* wq = (const float*)d_in[1];
  const float* bq = (const float*)d_in[2];
  const float* wk = (const float*)d_in[3];
  const float* bk = (const float*)d_in[4];
  const float* wv = (const float*)d_in[5];
  const float* bv = (const float*)d_in[6];
  const float* wo = (const float*)d_in[7];
  const float* bo = (const float*)d_in[8];
  float* out = (float*)d_out;

  // Workspace (f16): Q | K | V | AO (each 8M elems) | wq|wk|wv|wo (64K each)
  const size_t qkv = (size_t)BATCH * NTOK * CH;
  _Float16* Qf  = (_Float16*)d_ws;
  _Float16* Kf  = Qf + qkv;
  _Float16* Vf  = Kf + qkv;
  _Float16* AO  = Vf + qkv;
  _Float16* W16 = AO + qkv;            // 4 x 65536
  _Float16* wqh = W16;
  _Float16* wkh = W16 + 65536;
  _Float16* wvh = W16 + 2 * 65536;
  _Float16* woh = W16 + 3 * 65536;

  cvt_weights_kernel<<<64, 256, 0, stream>>>(wq, wk, wv, wo, W16);

  qkv_proj_kernel<<<1024, 256, 0, stream>>>(x, wqh, wkh, wvh, bq, bk, bv,
                                            Qf, Kf, Vf);

  const size_t smem = (size_t)(64 * KV_LD + 256 * VT_LD + 8 * 16 * P_LD)
                      * sizeof(_Float16);
  flash_attn_kernel<<<256, 256, smem, stream>>>(Qf, Kf, Vf, AO);

  out_proj_kernel<<<512, 256, 0, stream>>>(AO, woh, bo, out);
}
